// Attention_51926154608748
// MI455X (gfx1250) — matrix operand order repaired and checked by reading, not compiled, not run on hardware
//
#include <hip/hip_runtime.h>
#include <hip/hip_bf16.h>

typedef __attribute__((ext_vector_type(16))) __bf16 v16bf;
typedef __attribute__((ext_vector_type(8)))  __bf16 v8bf;
typedef __attribute__((ext_vector_type(8)))  float  v8f;

#define S_DIM 1024
#define D_DIM 1024
#define B_DIM 8
#define PSTR  (1024 + 16)   // padded LDS row stride (bf16 elements)

// ---------------------------------------------------------------------------
// A-operand loader: 16x32 bf16 tile from row-major f32.
// Lane L: row = L%16; elems 0..15 = K[kA .. kA+15]
// where kA = k0 + (L/16)*16.
// ---------------------------------------------------------------------------
__device__ __forceinline__ v16bf load_a_f32(const float* __restrict__ rowp, int kA) {
    v8f c0 = *(const v8f*)(rowp + kA);
    v8f c1 = *(const v8f*)(rowp + kA + 8);
    v16bf a;
#pragma unroll
    for (int i = 0; i < 8; ++i) {
        a[i]     = (__bf16)c0[i];
        a[i + 8] = (__bf16)c1[i];
    }
    return a;
}

// Same A layout but sourced from bf16 LDS (P matrix in phase 3).
__device__ __forceinline__ v16bf load_a_lds(const __bf16* prow, int kA) {
    v8bf c0 = *(const v8bf*)(prow + kA);
    v8bf c1 = *(const v8bf*)(prow + kA + 8);
    return __builtin_shufflevector(c0, c1, 0,1,2,3,4,5,6,7,8,9,10,11,12,13,14,15);
}

// ---------------------------------------------------------------------------
// 32x32 tiled transpose + f32->bf16 convert (per-matrix batched via blockIdx.z)
// ---------------------------------------------------------------------------
__global__ __launch_bounds__(256) void transpose_cvt(const float* __restrict__ src,
                                                     __bf16* __restrict__ dst) {
    __shared__ float tile[32][33];
    const size_t mat = (size_t)blockIdx.z * S_DIM * D_DIM;
    const int bx = blockIdx.x * 32, by = blockIdx.y * 32;
    const int tx = threadIdx.x, ty = threadIdx.y;   // block (32,8)
#pragma unroll
    for (int i = ty; i < 32; i += 8)
        tile[i][tx] = src[mat + (size_t)(by + i) * D_DIM + bx + tx];
    __syncthreads();
#pragma unroll
    for (int i = ty; i < 32; i += 8)
        dst[mat + (size_t)(bx + i) * S_DIM + by + tx] = (__bf16)tile[tx][i];
}

// ---------------------------------------------------------------------------
// GEMM1: py[b] = y[b] @ W + bias ; stored TRANSPOSED bf16: pyT[b][e][s].
// Grid (S/32, B); 32 rows per workgroup = 2 M-tiles per wave sharing B tiles.
// ---------------------------------------------------------------------------
__global__ __launch_bounds__(256) void gemm1_kernel(const float* __restrict__ y,
                                                    const __bf16* __restrict__ WT,
                                                    const float* __restrict__ bias,
                                                    __bf16* __restrict__ pyT) {
    const int b    = blockIdx.y;
    const int r0   = blockIdx.x * 32;
    const int lane = threadIdx.x & 31;
    const int wave = threadIdx.x >> 5;
    const int half = lane >> 4;
    const int lmod = lane & 15;

    v8f acc[2][8];
#pragma unroll
    for (int mi = 0; mi < 2; ++mi)
#pragma unroll
        for (int t = 0; t < 8; ++t)
#pragma unroll
            for (int r = 0; r < 8; ++r) acc[mi][t][r] = 0.0f;

    const float* yrow0 = y + ((size_t)b * S_DIM + r0 + lmod) * D_DIM;
    const float* yrow1 = yrow0 + 16 * D_DIM;
    const __bf16* Wbase = WT + (size_t)(wave * 128 + lmod) * D_DIM + half * 16;

    for (int k0 = 0; k0 < D_DIM; k0 += 32) {
        const v16bf a0 = load_a_f32(yrow0, k0 + half * 16);
        const v16bf a1 = load_a_f32(yrow1, k0 + half * 16);
        v16bf bm = *(const v16bf*)(Wbase + k0);
#pragma unroll
        for (int t = 0; t < 8; ++t) {
            v16bf bnext = bm;
            if (t < 7) bnext = *(const v16bf*)(Wbase + (size_t)(t + 1) * 16 * D_DIM + k0);
            acc[0][t] = __builtin_amdgcn_wmma_f32_16x16x32_bf16(
                false, a0, false, bm, (short)0, acc[0][t], false, false);
            acc[1][t] = __builtin_amdgcn_wmma_f32_16x16x32_bf16(
                false, a1, false, bm, (short)0, acc[1][t], false, false);
            bm = bnext;
        }
    }

#pragma unroll
    for (int t = 0; t < 8; ++t) {
        const int n = wave * 128 + t * 16 + lmod;
        const float bv = bias[n];
#pragma unroll
        for (int mi = 0; mi < 2; ++mi) {
            v8bf pk;
#pragma unroll
            for (int r = 0; r < 8; ++r) pk[r] = (__bf16)(acc[mi][t][r] + bv);
            *(v8bf*)(pyT + ((size_t)b * D_DIM + n) * S_DIM + r0 + mi * 16 + half * 8) = pk;
        }
    }
}

// ---------------------------------------------------------------------------
// Fused attention: S = x @ py  (+mask*-1e9) -> row softmax -> out = P @ x.
// Grid (S/32, B); 32 rows per workgroup, 1024 cols split over 8 waves.
// ---------------------------------------------------------------------------
__global__ __launch_bounds__(256) void attn_kernel(const float* __restrict__ x,
                                                   const __bf16* __restrict__ pyT,
                                                   const float* __restrict__ mask,
                                                   const __bf16* __restrict__ xT,
                                                   float* __restrict__ out) {
    __shared__ __bf16 ldsP[32 * PSTR];   // P (32 x 1024) bf16, padded rows
    __shared__ float  red[8][32];        // cross-wave reduction scratch

    const int b    = blockIdx.y;
    const int r0   = blockIdx.x * 32;
    const int lane = threadIdx.x & 31;
    const int wave = threadIdx.x >> 5;
    const int half = lane >> 4;
    const int lmod = lane & 15;

    v8f acc[2][8];
#pragma unroll
    for (int mi = 0; mi < 2; ++mi)
#pragma unroll
        for (int t = 0; t < 8; ++t)
#pragma unroll
            for (int r = 0; r < 8; ++r) acc[mi][t][r] = 0.0f;

    // ---- Phase 1: S-tile = x_rows @ py  (K = 1024) -----------------------
    {
        const float* xrow0 = x + ((size_t)b * S_DIM + r0 + lmod) * D_DIM;
        const float* xrow1 = xrow0 + 16 * D_DIM;
        const __bf16* Bbase = pyT + ((size_t)b * D_DIM + wave * 128 + lmod) * S_DIM + half * 16;
        for (int k0 = 0; k0 < D_DIM; k0 += 32) {
            const v16bf a0 = load_a_f32(xrow0, k0 + half * 16);
            const v16bf a1 = load_a_f32(xrow1, k0 + half * 16);
            v16bf bm = *(const v16bf*)(Bbase + k0);
#pragma unroll
            for (int t = 0; t < 8; ++t) {
                v16bf bnext = bm;
                if (t < 7) bnext = *(const v16bf*)(Bbase + (size_t)(t + 1) * 16 * S_DIM + k0);
                acc[0][t] = __builtin_amdgcn_wmma_f32_16x16x32_bf16(
                    false, a0, false, bm, (short)0, acc[0][t], false, false);
                acc[1][t] = __builtin_amdgcn_wmma_f32_16x16x32_bf16(
                    false, a1, false, bm, (short)0, acc[1][t], false, false);
                bm = bnext;
            }
        }
    }

    // ---- Phase 2: mask + row softmax over distributed 32x1024 ------------
    float lm[2][8];
#pragma unroll
    for (int mi = 0; mi < 2; ++mi)
#pragma unroll
        for (int r = 0; r < 8; ++r) lm[mi][r] = -3.4e38f;

#pragma unroll
    for (int t = 0; t < 8; ++t) {
        const int col = wave * 128 + t * 16 + lmod;
#pragma unroll
        for (int mi = 0; mi < 2; ++mi)
#pragma unroll
            for (int r = 0; r < 8; ++r) {
                const int row = r0 + mi * 16 + half * 8 + r;
                const float mv = __builtin_nontemporal_load(
                    &mask[((size_t)b * S_DIM + row) * S_DIM + col]);
                const float v = acc[mi][t][r] + mv * -1e9f;
                acc[mi][t][r] = v;
                lm[mi][r] = fmaxf(lm[mi][r], v);
            }
    }
#pragma unroll
    for (int off = 1; off < 16; off <<= 1)
#pragma unroll
        for (int mi = 0; mi < 2; ++mi)
#pragma unroll
            for (int r = 0; r < 8; ++r)
                lm[mi][r] = fmaxf(lm[mi][r], __shfl_xor(lm[mi][r], off));
    if (lmod == 0) {
#pragma unroll
        for (int mi = 0; mi < 2; ++mi)
#pragma unroll
            for (int r = 0; r < 8; ++r) red[wave][mi * 16 + half * 8 + r] = lm[mi][r];
    }
    __syncthreads();
    float rm[2][8];
#pragma unroll
    for (int mi = 0; mi < 2; ++mi)
#pragma unroll
        for (int r = 0; r < 8; ++r) {
            float m = -3.4e38f;
#pragma unroll
            for (int ww = 0; ww < 8; ++ww) m = fmaxf(m, red[ww][mi * 16 + half * 8 + r]);
            rm[mi][r] = m;
        }
    __syncthreads();

    float ls[2][8];
#pragma unroll
    for (int mi = 0; mi < 2; ++mi)
#pragma unroll
        for (int r = 0; r < 8; ++r) ls[mi][r] = 0.0f;
#pragma unroll
    for (int t = 0; t < 8; ++t)
#pragma unroll
        for (int mi = 0; mi < 2; ++mi)
#pragma unroll
            for (int r = 0; r < 8; ++r) {
                const float e = __expf(acc[mi][t][r] - rm[mi][r]);
                acc[mi][t][r] = e;
                ls[mi][r] += e;
            }
#pragma unroll
    for (int off = 1; off < 16; off <<= 1)
#pragma unroll
        for (int mi = 0; mi < 2; ++mi)
#pragma unroll
            for (int r = 0; r < 8; ++r) ls[mi][r] += __shfl_xor(ls[mi][r], off);
    if (lmod == 0) {
#pragma unroll
        for (int mi = 0; mi < 2; ++mi)
#pragma unroll
            for (int r = 0; r < 8; ++r) red[wave][mi * 16 + half * 8 + r] = ls[mi][r];
    }
    __syncthreads();
    float inv[2][8];
#pragma unroll
    for (int mi = 0; mi < 2; ++mi)
#pragma unroll
        for (int r = 0; r < 8; ++r) {
            float s = 0.0f;
#pragma unroll
            for (int ww = 0; ww < 8; ++ww) s += red[ww][mi * 16 + half * 8 + r];
            inv[mi][r] = 1.0f / s;
        }

    // stage normalized P (bf16) into LDS, row-major with padding
#pragma unroll
    for (int t = 0; t < 8; ++t) {
        const int col = wave * 128 + t * 16 + lmod;
#pragma unroll
        for (int mi = 0; mi < 2; ++mi)
#pragma unroll
            for (int r = 0; r < 8; ++r) {
                const int m = mi * 16 + half * 8 + r;
                ldsP[m * PSTR + col] = (__bf16)(acc[mi][t][r] * inv[mi][r]);
            }
    }
    __syncthreads();

    // ---- Phase 3: out = P @ x  (K = 1024, B-operand from xT) -------------
#pragma unroll
    for (int mi = 0; mi < 2; ++mi)
#pragma unroll
        for (int t = 0; t < 8; ++t)
#pragma unroll
            for (int r = 0; r < 8; ++r) acc[mi][t][r] = 0.0f;

    {
        const __bf16* prow0 = ldsP + lmod * PSTR;
        const __bf16* prow1 = ldsP + (16 + lmod) * PSTR;
        const __bf16* Bbase = xT + ((size_t)b * D_DIM + wave * 128 + lmod) * S_DIM + half * 16;
        for (int k0 = 0; k0 < S_DIM; k0 += 32) {
            const v16bf a0 = load_a_lds(prow0, k0 + half * 16);
            const v16bf a1 = load_a_lds(prow1, k0 + half * 16);
            v16bf bm = *(const v16bf*)(Bbase + k0);
#pragma unroll
            for (int t = 0; t < 8; ++t) {
                v16bf bnext = bm;
                if (t < 7) bnext = *(const v16bf*)(Bbase + (size_t)(t + 1) * 16 * S_DIM + k0);
                acc[0][t] = __builtin_amdgcn_wmma_f32_16x16x32_bf16(
                    false, a0, false, bm, (short)0, acc[0][t], false, false);
                acc[1][t] = __builtin_amdgcn_wmma_f32_16x16x32_bf16(
                    false, a1, false, bm, (short)0, acc[1][t], false, false);
                bm = bnext;
            }
        }
    }

#pragma unroll
    for (int t = 0; t < 8; ++t) {
        const int col = wave * 128 + t * 16 + lmod;
#pragma unroll
        for (int mi = 0; mi < 2; ++mi)
#pragma unroll
            for (int r = 0; r < 8; ++r) {
                const int row = r0 + mi * 16 + half * 8 + r;
                __builtin_nontemporal_store(acc[mi][t][r],
                    &out[((size_t)b * S_DIM + row) * D_DIM + col]);
            }
    }
}

// ---------------------------------------------------------------------------
// Launch: inputs (setup_inputs order): x, y, mask, W, b
// ws layout (bf16): WT [1024*1024] | xT [8*1024*1024] | pyT [8*1024*1024]
// ---------------------------------------------------------------------------
extern "C" void kernel_launch(void* const* d_in, const int* in_sizes, int n_in,
                              void* d_out, int out_size, void* d_ws, size_t ws_size,
                              hipStream_t stream) {
    const float* x    = (const float*)d_in[0];
    const float* y    = (const float*)d_in[1];
    const float* mask = (const float*)d_in[2];
    const float* W    = (const float*)d_in[3];
    const float* bias = (const float*)d_in[4];
    float* out = (float*)d_out;

    __bf16* WT  = (__bf16*)d_ws;
    __bf16* xT  = WT + (size_t)S_DIM * D_DIM;
    __bf16* pyT = xT + (size_t)B_DIM * S_DIM * D_DIM;

    dim3 tb(32, 8);
    transpose_cvt<<<dim3(32, 32, 1),     tb, 0, stream>>>(W, WT);
    transpose_cvt<<<dim3(32, 32, B_DIM), tb, 0, stream>>>(x, xT);

    gemm1_kernel<<<dim3(S_DIM / 32, B_DIM), 256, 0, stream>>>(y, WT, bias, pyT);
    attn_kernel <<<dim3(S_DIM / 32, B_DIM), 256, 0, stream>>>(x, pyT, mask, xT, out);
}
